// MultiHeadAttention_30855045054734
// MI455X (gfx1250) — matrix operand order repaired and checked by reading, not compiled, not run on hardware
//
#include <hip/hip_runtime.h>
#include <hip/hip_bf16.h>

// ---------------------------------------------------------------------------
// Causal multi-head attention forward, MI455X (gfx1250, wave32, WMMA).
//   B=2, S=2048, D=1024, H=16, HD=64.
// Pipeline:
//   cvt_x    : x f32 -> f16 (once; feeds 3 GEMMs)
//   pack_w   : Wq/Wk/Wv/Wo f32 -> f16 B-fragment-major (one 32B read/lane/frag)
//   qkv_gemm : xh @ [Wq|Wk|Wv], 32x64 tile/wave, 8 WMMA per k-step
//   flash    : online-softmax attention, 64-key chunks, 16 WMMA per chunk
//   out_gemm : ctx @ Wo + bo -> f32
// ---------------------------------------------------------------------------

constexpr int BATCH  = 2;
constexpr int SEQ    = 2048;
constexpr int DMODEL = 1024;
constexpr int NH     = 16;
constexpr int HDIM   = 64;
constexpr int NT16   = DMODEL / 16;   // 16-col tiles per weight matrix (64)

typedef _Float16 h8   __attribute__((ext_vector_type(8)));
typedef _Float16 v16h __attribute__((ext_vector_type(16)));
typedef float    v8f  __attribute__((ext_vector_type(8)));

__device__ __forceinline__ v8f wmma16(v16h a, v16h b, v8f c) {
  return __builtin_amdgcn_wmma_f32_16x16x32_f16(
      false, a, false, b, (short)0, c, false, false);
}

__device__ __forceinline__ v16h pack16(h8 lo, h8 hi) {
  v16h r;
#pragma unroll
  for (int i = 0; i < 8; ++i) { r[i] = lo[i]; r[8 + i] = hi[i]; }
  return r;
}

// ---------------------------------------------------------------------------
// x f32 -> f16, 8 elements/thread, b128 in / b128 out.
// ---------------------------------------------------------------------------
__global__ __launch_bounds__(256) void cvt_x(const float* __restrict__ x,
                                             _Float16* __restrict__ xh) {
  const size_t idx = ((size_t)blockIdx.x * 256 + threadIdx.x) * 8;
  const float4 a = *(const float4*)(x + idx);
  const float4 b = *(const float4*)(x + idx + 4);
  h8 o;
  o[0] = (_Float16)a.x; o[1] = (_Float16)a.y;
  o[2] = (_Float16)a.z; o[3] = (_Float16)a.w;
  o[4] = (_Float16)b.x; o[5] = (_Float16)b.y;
  o[6] = (_Float16)b.z; o[7] = (_Float16)b.w;
  *(h8*)(xh + idx) = o;
}

// ---------------------------------------------------------------------------
// Pack W [D x D] f32 into f16 B-fragment-major layout:
//   wp[((kt*NT16 + ntile)*32 + lane)*16 + i]
//     = W[kt*32 + 16*(lane>>4) + i][ntile*16 + (lane&15)]
// so a 32x16 WMMA B fragment is one contiguous 32B read per lane.
// ---------------------------------------------------------------------------
__global__ __launch_bounds__(256) void pack_w(const float* __restrict__ W,
                                              _Float16* __restrict__ wp) {
  const int wid  = blockIdx.x * 8 + (threadIdx.x >> 5);  // tile index
  const int lane = threadIdx.x & 31;
  const int ntile = wid % NT16;
  const int kt    = wid / NT16;
  const int k0 = kt * 32 + 16 * (lane >> 4);
  const int n  = ntile * 16 + (lane & 15);
  h8 v0, v1;
#pragma unroll
  for (int i = 0; i < 8; ++i) v0[i] = (_Float16)W[(size_t)(k0 + i) * DMODEL + n];
#pragma unroll
  for (int i = 0; i < 8; ++i) v1[i] = (_Float16)W[(size_t)(k0 + 8 + i) * DMODEL + n];
  _Float16* dst = wp + ((size_t)wid * 32 + lane) * 16;
  *(h8*)dst = v0;
  *(h8*)(dst + 8) = v1;
}

// ---------------------------------------------------------------------------
// Core 32x64-tile f16 GEMM accumulation (A row-major f16, B packed).
// 8 WMMAs per 32-wide k-step; A frags: 2x (two b128 loads), B frags: 4x v16h.
// ---------------------------------------------------------------------------
__device__ __forceinline__ void gemm_acc_32x64(
    const _Float16* __restrict__ A, const _Float16* __restrict__ Bp,
    int r0, int n0t, int lane, v8f c[2][4]) {
  const int hi = lane >> 4, mr = lane & 15;
  for (int k0 = 0; k0 < DMODEL; k0 += 32) {
    v16h a[2];
#pragma unroll
    for (int mi = 0; mi < 2; ++mi) {
      const _Float16* ar = A + (size_t)(r0 + 16 * mi + mr) * DMODEL + k0 + 16 * hi;
      a[mi] = pack16(*(const h8*)ar, *(const h8*)(ar + 8));
    }
#pragma unroll
    for (int ni = 0; ni < 4; ++ni) {
      const v16h b = *(const v16h*)(
          Bp + ((size_t)((k0 >> 5) * NT16 + n0t + ni) * 32 + lane) * 16);
      c[0][ni] = wmma16(a[0], b, c[0][ni]);
      c[1][ni] = wmma16(a[1], b, c[1][ni]);
    }
  }
}

// ---------------------------------------------------------------------------
// Kernel: QKV projection. blockIdx.y in [0,48): mat = y/16, head = y%16.
// Q,K stored f16 [B,H,S,HD]; V stored transposed f16 [B,H,HD,S].
// ---------------------------------------------------------------------------
__global__ __launch_bounds__(32) void qkv_gemm(
    const _Float16* __restrict__ xh,
    const _Float16* __restrict__ wqp, const _Float16* __restrict__ wkp,
    const _Float16* __restrict__ wvp,
    _Float16* __restrict__ qws, _Float16* __restrict__ kws,
    _Float16* __restrict__ vtws) {
  const int lane = threadIdx.x & 31;
  const int hi = lane >> 4, mr = lane & 15;
  const int r0 = blockIdx.x * 32;
  const int mat = blockIdx.y >> 4;        // 0=Q,1=K,2=V (uniform)
  const int h   = blockIdx.y & 15;        // head (uniform)
  const int n0t = h * 4;                  // 16-col tile index within matrix

  const _Float16* Bp = (mat == 0) ? wqp : ((mat == 1) ? wkp : wvp);
  v8f c[2][4] = {{{}, {}, {}, {}}, {{}, {}, {}, {}}};
  gemm_acc_32x64(xh, Bp, r0, n0t, lane, c);

  if (mat != 2) {
    _Float16* dst = (mat == 0) ? qws : kws;
#pragma unroll
    for (int mi = 0; mi < 2; ++mi)
#pragma unroll
      for (int j = 0; j < 8; ++j) {
        const int row = r0 + 16 * mi + j + 8 * hi;
        const int bb = row >> 11, s = row & (SEQ - 1);
        _Float16* p = dst + (((size_t)bb * NH + h) * SEQ + s) * HDIM;
#pragma unroll
        for (int ni = 0; ni < 4; ++ni)
          p[16 * ni + mr] = (_Float16)c[mi][ni][j];
      }
  } else {
#pragma unroll
    for (int mi = 0; mi < 2; ++mi)
#pragma unroll
      for (int j = 0; j < 8; ++j) {
        const int row = r0 + 16 * mi + j + 8 * hi;
        const int bb = row >> 11, s = row & (SEQ - 1);
        _Float16* p = vtws + ((size_t)bb * NH + h) * HDIM * SEQ + s;
#pragma unroll
        for (int ni = 0; ni < 4; ++ni)
          p[(size_t)(16 * ni + mr) * SEQ] = (_Float16)c[mi][ni][j];
      }
  }
}

// ---------------------------------------------------------------------------
// Kernel: causal flash attention. One wave per (b*h, 16-query tile).
// 64-key chunks: 8 WMMA for S=Q.K^T, base-2 online softmax, LDS transpose of
// P (C-layout f32 -> A-layout f16), 8 WMMA for O += P.V.
// ---------------------------------------------------------------------------
__global__ __launch_bounds__(32) void flash_attn(
    const _Float16* __restrict__ qws, const _Float16* __restrict__ kws,
    const _Float16* __restrict__ vtws, _Float16* __restrict__ ctx) {
  __shared__ _Float16 plds[16 * 72];  // 16 rows x 64 cols, stride 72

  const int lane = threadIdx.x & 31;
  const int hi = lane >> 4, mr = lane & 15;
  const int q0 = blockIdx.x * 16;
  const int bh = blockIdx.y;
  const int bt = bh / NH, hh = bh % NH;

  const _Float16* Qb = qws  + (size_t)bh * SEQ * HDIM;
  const _Float16* Kb = kws  + (size_t)bh * SEQ * HDIM;
  const _Float16* Vt = vtws + (size_t)bh * HDIM * SEQ;

  // Q A-fragments (hd 0..31, 32..63), held across the key loop.
  v16h qa[2];
#pragma unroll
  for (int f = 0; f < 2; ++f) {
    const _Float16* qr = Qb + (size_t)(q0 + mr) * HDIM + 32 * f + 16 * hi;
    qa[f] = pack16(*(const h8*)qr, *(const h8*)(qr + 8));
  }

  v8f o[4] = {{}, {}, {}, {}};
  float mrow[8], lrow[8];
#pragma unroll
  for (int j = 0; j < 8; ++j) { mrow[j] = -1.0e30f; lrow[j] = 0.0f; }

  // scores kept in base-2 domain: s2 = raw * (1/sqrt(64)) * log2(e)
  const float C = 0.125f * 1.44269504088896340736f;
  const int kend = q0 + 16;

  for (int kc = 0; kc < kend; kc += 64) {
    // ---- scores: four 16-key tiles ----
    v8f st[4];
#pragma unroll
    for (int t = 0; t < 4; ++t) {
      v8f sc = {};
#pragma unroll
      for (int f = 0; f < 2; ++f) {
        const v16h kb = *(const v16h*)(
            Kb + (size_t)(kc + 16 * t + mr) * HDIM + 32 * f + 16 * hi);
        sc = wmma16(qa[f], kb, sc);
      }
      st[t] = sc;
    }

    // ---- scale (base-2) + causal mask ----
    const bool need_mask = (kc + 63 >= q0);
#pragma unroll
    for (int t = 0; t < 4; ++t)
#pragma unroll
      for (int j = 0; j < 8; ++j) {
        float a = st[t][j] * C;
        if (need_mask && (kc + 16 * t + mr > q0 + j + 8 * hi)) a = -1.0e30f;
        st[t][j] = a;
      }

    // ---- online softmax (base 2); rows live in 16-lane halves ----
#pragma unroll
    for (int j = 0; j < 8; ++j) {
      float mx = fmaxf(fmaxf(st[0][j], st[1][j]), fmaxf(st[2][j], st[3][j]));
#pragma unroll
      for (int off = 8; off >= 1; off >>= 1)
        mx = fmaxf(mx, __shfl_xor(mx, off, 16));
      const float newm  = fmaxf(mrow[j], mx);
      const float alpha = exp2f(mrow[j] - newm);
      float p[4];
#pragma unroll
      for (int t = 0; t < 4; ++t) p[t] = exp2f(st[t][j] - newm);
      float rs = (p[0] + p[1]) + (p[2] + p[3]);
#pragma unroll
      for (int off = 8; off >= 1; off >>= 1) rs += __shfl_xor(rs, off, 16);
      lrow[j] = lrow[j] * alpha + rs;
      mrow[j] = newm;
      o[0][j] *= alpha; o[1][j] *= alpha; o[2][j] *= alpha; o[3][j] *= alpha;
      const int mm = j + 8 * hi;
#pragma unroll
      for (int t = 0; t < 4; ++t)
        plds[mm * 72 + 16 * t + mr] = (_Float16)p[t];
    }
    __syncthreads();  // single-wave WG: LDS fence; barrier itself is a NOP

    // ---- P as A-fragments (keys kc..kc+31, kc+32..kc+63) ----
    v16h pa[2];
#pragma unroll
    for (int f = 0; f < 2; ++f) {
      const _Float16* pr = plds + mr * 72 + 32 * f + 16 * hi;
      pa[f] = pack16(*(const h8*)pr, *(const h8*)(pr + 8));
    }

    // ---- O += P * V over 4 hd tiles ----
#pragma unroll
    for (int f = 0; f < 2; ++f)
#pragma unroll
      for (int nt = 0; nt < 4; ++nt) {
        const v16h vb = *(const v16h*)(
            Vt + (size_t)(16 * nt + mr) * SEQ + kc + 32 * f + 16 * hi);
        o[nt] = wmma16(pa[f], vb, o[nt]);
      }
    __syncthreads();  // protect plds before next chunk
  }

  // ---- normalize, write ctx f16 [B, S, H*HD] ----
#pragma unroll
  for (int j = 0; j < 8; ++j) {
    const float inv = 1.0f / lrow[j];
    const int s = q0 + j + 8 * hi;
    _Float16* dst = ctx + ((size_t)bt * SEQ + s) * DMODEL + hh * HDIM;
#pragma unroll
    for (int nt = 0; nt < 4; ++nt)
      dst[16 * nt + mr] = (_Float16)(o[nt][j] * inv);
  }
}

// ---------------------------------------------------------------------------
// Kernel: output projection. out = ctx @ Wo + bo (f32 out).
// ---------------------------------------------------------------------------
__global__ __launch_bounds__(32) void out_gemm(
    const _Float16* __restrict__ ctx, const _Float16* __restrict__ wop,
    const float* __restrict__ bo, float* __restrict__ out) {
  const int lane = threadIdx.x & 31;
  const int hi = lane >> 4, mr = lane & 15;
  const int r0 = blockIdx.x * 32;
  const int n0t = blockIdx.y * 4;

  v8f c[2][4] = {{{}, {}, {}, {}}, {{}, {}, {}, {}}};
  gemm_acc_32x64(ctx, wop, r0, n0t, lane, c);

  float bias[4];
#pragma unroll
  for (int ni = 0; ni < 4; ++ni) bias[ni] = bo[(n0t + ni) * 16 + mr];

#pragma unroll
  for (int mi = 0; mi < 2; ++mi)
#pragma unroll
    for (int j = 0; j < 8; ++j) {
      float* p = out + (size_t)(r0 + 16 * mi + j + 8 * hi) * DMODEL;
#pragma unroll
      for (int ni = 0; ni < 4; ++ni)
        p[(n0t + ni) * 16 + mr] = c[mi][ni][j] + bias[ni];
    }
}

// ---------------------------------------------------------------------------
extern "C" void kernel_launch(void* const* d_in, const int* in_sizes, int n_in,
                              void* d_out, int out_size, void* d_ws,
                              size_t ws_size, hipStream_t stream) {
  (void)in_sizes; (void)n_in; (void)out_size; (void)ws_size;
  const float* x  = (const float*)d_in[0];
  const float* Wq = (const float*)d_in[1];
  const float* Wk = (const float*)d_in[2];
  const float* Wv = (const float*)d_in[3];
  const float* Wo = (const float*)d_in[4];
  const float* bo = (const float*)d_in[5];
  float* out = (float*)d_out;

  // Workspace (f16): xh 8MB | wq/wk/wv/wo packed 4x2MB | Q,K,V^T 3x8MB |
  // ctx 8MB  => 48 MB total.
  char* ws = (char*)d_ws;
  const size_t xb  = (size_t)BATCH * SEQ * DMODEL * 2;     // 8 MB
  const size_t wb  = (size_t)DMODEL * DMODEL * 2;          // 2 MB
  const size_t seg = (size_t)BATCH * NH * SEQ * HDIM * 2;  // 8 MB
  _Float16* xh   = (_Float16*)(ws);
  _Float16* wqp  = (_Float16*)(ws + xb);
  _Float16* wkp  = (_Float16*)(ws + xb + wb);
  _Float16* wvp  = (_Float16*)(ws + xb + 2 * wb);
  _Float16* wop  = (_Float16*)(ws + xb + 3 * wb);
  _Float16* qws  = (_Float16*)(ws + xb + 4 * wb);
  _Float16* kws  = (_Float16*)(ws + xb + 4 * wb + seg);
  _Float16* vtws = (_Float16*)(ws + xb + 4 * wb + 2 * seg);
  _Float16* ctx  = (_Float16*)(ws + xb + 4 * wb + 3 * seg);

  cvt_x<<<(BATCH * SEQ * DMODEL) / (256 * 8), 256, 0, stream>>>(x, xh);
  const int packBlocks = (DMODEL / 32) * NT16 / 8;  // 2048 tiles / 8 per block
  pack_w<<<packBlocks, 256, 0, stream>>>(Wq, wqp);
  pack_w<<<packBlocks, 256, 0, stream>>>(Wk, wkp);
  pack_w<<<packBlocks, 256, 0, stream>>>(Wv, wvp);
  pack_w<<<packBlocks, 256, 0, stream>>>(Wo, wop);

  qkv_gemm<<<dim3((BATCH * SEQ) / 32, 48), 32, 0, stream>>>(
      xh, wqp, wkp, wvp, qws, kws, vtws);
  flash_attn<<<dim3(SEQ / 16, BATCH * NH), 32, 0, stream>>>(qws, kws, vtws,
                                                            ctx);
  out_gemm<<<dim3((BATCH * SEQ) / 32, DMODEL / 64), 32, 0, stream>>>(
      ctx, wop, bo, out);
}
